// ConvOffset2d_41704132444288
// MI455X (gfx1250) — compile-verified, matrix operand order checked by reading, not run on hardware
//
#include <hip/hip_runtime.h>

// ---------------------------------------------------------------------------
// Deformable conv2d (B=8, Cin=64, H=W=128, Cout=64, G=4, CPG=16, 3x3, pad=1)
// Strategy: fused bilinear-gather im2col -> bf16 WMMA GEMM on CDNA5 (gfx1250).
//   M = Cout = 64, N = 64 pixels per block, Kred = Cin*9 = 576 (18 chunks of 32)
//   v_wmma_f32_16x16x32_bf16, f32 accumulation.
// ---------------------------------------------------------------------------

typedef __attribute__((ext_vector_type(16))) __bf16 v16bf;
typedef __attribute__((ext_vector_type(8)))  float  v8f;

#define BATCH  8
#define CIN    64
#define HH     128
#define WW     128
#define COUT   64
#define GRP    4
#define CPG_   16
#define KPTS   9          // 3x3
#define KRED   576        // CIN * 9
#define KCH    18         // KRED / 32
#define NTIL   64         // pixels per workgroup
#define HW     (HH * WW)  // 16384

__device__ __forceinline__ __bf16 f2bf(float f) {
    unsigned u = __builtin_bit_cast(unsigned, f);
    unsigned r = u + 0x7FFFu + ((u >> 16) & 1u);   // round-to-nearest-even
    unsigned short h = (unsigned short)(r >> 16);
    return __builtin_bit_cast(__bf16, h);
}

__device__ __forceinline__ int imin(int a, int b) { return a < b ? a : b; }
__device__ __forceinline__ int imax(int a, int b) { return a > b ? a : b; }

// Swizzled LDS layout for WMMA bf16 operands: per (k-chunk, 16-wide tile) a
// wave reads 16 contiguous bf16 per lane (32B, aligned) -> direct v16bf load.
// Element index e within a lane for K-index kk (0..31), lane half = (kk>>3)&1:
//   e = ((kk>>4)<<3) | (kk&7)
__device__ __forceinline__ int swz(int kc, int tile, int lane) {
    return (((kc * 4 + tile) * 32 + lane) << 4);
}

extern "C" __global__ __launch_bounds__(256)
void deform_conv_wmma(const float* __restrict__ x,
                      const float* __restrict__ offset,
                      const float* __restrict__ weight,
                      float* __restrict__ out)
{
    extern __shared__ char lds_raw[];
    __bf16* Alds = (__bf16*)lds_raw;        // 64 x 576 weight, swizzled (73728 B)
    __bf16* Slds = Alds + COUT * KRED;      // 576 x 64 sampled, swizzled (73728 B)

    const int tid = threadIdx.x;
    const int wg  = blockIdx.x;             // 0 .. 2047
    const int b   = wg >> 8;                // / 256
    const int rem = wg & 255;
    const int ho  = rem >> 1;
    const int wo0 = (rem & 1) << 6;         // 0 or 64

    // ---- Phase 1: weight (Cout x Kred) f32 -> bf16, swizzled A-operand ----
    for (int i = tid; i < COUT * KRED; i += 256) {
        int m = i / KRED;
        int r = i - m * KRED;
        int kc = r >> 5, kk = r & 31;
        int lane = (m & 15) | (((kk >> 3) & 1) << 4);
        int e    = ((kk >> 4) << 3) | (kk & 7);
        Alds[swz(kc, m >> 4, lane) + e] = f2bf(weight[i]);
    }

    // ---- Phase 2: bilinear gather -> swizzled B-operand (576 x 64) ----
    const float* xb = x      + (size_t)b * CIN * HW;
    const float* ob = offset + (size_t)b * (GRP * KPTS * 2) * HW;

    for (int loc = tid; loc < NTIL * GRP * KPTS; loc += 256) {
        int n  = loc & 63;
        int gk = loc >> 6;                  // 0..35
        int g  = gk / KPTS;
        int k  = gk - g * KPTS;
        int ky = k / 3, kx = k - ky * 3;
        int wo = wo0 + n;

        int oidx = ((g * KPTS + k) * 2) * HW + ho * WW + wo;
        float dy = ob[oidx];
        float dx = ob[oidx + HW];
        float py = dy + (float)ky + (float)(ho - 1);    // stride 1, pad 1
        float px = dx + (float)kx + (float)(wo - 1);

        float y0f = floorf(py), x0f = floorf(px);
        int   y0 = (int)y0f,    x0 = (int)x0f;
        int   y1 = y0 + 1,      x1 = x0 + 1;
        float ly = py - y0f, hy = 1.0f - ly;
        float lx = px - x0f, hx = 1.0f - lx;
        float vy0 = (y0 >= 0 && y0 < HH) ? 1.0f : 0.0f;
        float vy1 = (y1 >= 0 && y1 < HH) ? 1.0f : 0.0f;
        float vx0 = (x0 >= 0 && x0 < WW) ? 1.0f : 0.0f;
        float vx1 = (x1 >= 0 && x1 < WW) ? 1.0f : 0.0f;
        float w00 = hy * hx * vy0 * vx0;
        float w01 = hy * lx * vy0 * vx1;
        float w10 = ly * hx * vy1 * vx0;
        float w11 = ly * lx * vy1 * vx1;

        int yc0 = imin(imax(y0, 0), HH - 1), yc1 = imin(imax(y1, 0), HH - 1);
        int xc0 = imin(imax(x0, 0), WW - 1), xc1 = imin(imax(x1, 0), WW - 1);
        int i00 = yc0 * WW + xc0, i01 = yc0 * WW + xc1;
        int i10 = yc1 * WW + xc0, i11 = yc1 * WW + xc1;

        const float* xg = xb + g * CPG_ * HW;
        int ntile = n >> 4;
        int ncol  = n & 15;
        #pragma unroll
        for (int c = 0; c < CPG_; ++c) {
            const float* p = xg + c * HW;
            float v = w00 * p[i00] + w01 * p[i01] + w10 * p[i10] + w11 * p[i11];
            int r  = (g * CPG_ + c) * KPTS + k;
            int kc = r >> 5, kk = r & 31;
            int lane = ncol | (((kk >> 3) & 1) << 4);
            int e    = ((kk >> 4) << 3) | (kk & 7);
            Slds[swz(kc, ntile, lane) + e] = f2bf(v);
        }
    }

    __syncthreads();

    // ---- Phase 3: WMMA GEMM. 8 waves x 2 (16x16) output tiles each ----
    const int lane = tid & 31;
    const int wave = tid >> 5;
    const int mi   = wave >> 1;             // 0..3  (Cout tile)
    const int ni0  = (wave & 1) << 1;       // 0 or 2 (pixel tiles ni0, ni0+1)

    v8f acc0 = {};
    v8f acc1 = {};
    #pragma unroll 3
    for (int kc = 0; kc < KCH; ++kc) {
        v16bf a  = *(const v16bf*)&Alds[swz(kc, mi,      lane)];
        v16bf b0 = *(const v16bf*)&Slds[swz(kc, ni0,     lane)];
        v16bf b1 = *(const v16bf*)&Slds[swz(kc, ni0 + 1, lane)];
        acc0 = __builtin_amdgcn_wmma_f32_16x16x32_bf16(false, a, false, b0,
                                                       (short)0, acc0, false, false);
        acc1 = __builtin_amdgcn_wmma_f32_16x16x32_bf16(false, a, false, b1,
                                                       (short)0, acc1, false, false);
    }

    // ---- Store D: VGPR j -> rows M=j (lanes 0-15) / M=j+8 (lanes 16-31) ----
    float* outb = out + (size_t)b * COUT * HW + (size_t)ho * WW + wo0;
    const int ncol = lane & 15;
    const int moff = (lane >> 4) << 3;
    #pragma unroll
    for (int j = 0; j < 8; ++j) {
        int o = mi * 16 + moff + j;
        outb[(size_t)o * HW +  ni0      * 16 + ncol] = acc0[j];
        outb[(size_t)o * HW + (ni0 + 1) * 16 + ncol] = acc1[j];
    }
}

extern "C" void kernel_launch(void* const* d_in, const int* in_sizes, int n_in,
                              void* d_out, int out_size, void* d_ws, size_t ws_size,
                              hipStream_t stream)
{
    const float* x      = (const float*)d_in[0];
    const float* offset = (const float*)d_in[1];
    const float* weight = (const float*)d_in[2];
    float*       out    = (float*)d_out;

    const size_t lds_bytes = (size_t)2 * COUT * KRED * sizeof(__bf16); // 147456

    // Opt in to >64KB dynamic LDS (WGP has 320KB). Attribute set is
    // idempotent/deterministic and not a stream operation.
    (void)hipFuncSetAttribute((const void*)deform_conv_wmma,
                              hipFuncAttributeMaxDynamicSharedMemorySize,
                              (int)lds_bytes);

    dim3 grid(BATCH * HH * (WW / NTIL));   // 8 * 128 * 2 = 2048 blocks
    dim3 block(256);                       // 8 wave32
    deform_conv_wmma<<<grid, block, lds_bytes, stream>>>(x, offset, weight, out);
}
